// SoftmaxRefMatcher_69672959475859
// MI455X (gfx1250) — matrix operand order repaired and checked by reading, not compiled, not run on hardware
//
#include <hip/hip_runtime.h>
#include <hip/hip_bf16.h>
#include <math.h>

// ---------------------------------------------------------------------------
// Problem constants (from the reference)
// ---------------------------------------------------------------------------
#define BATCH      2          // B
#define WSIZE      2          // window size
#define NKP        1024       // N keypoints
#define CDESC      248        // descriptor channels
#define KPAD       256        // C padded to multiple of 32 for WMMA K
#define WIDTH      256        // cart pixel width
#define MTOT       (WIDTH*WIDTH)   // 65536 dense positions
#define M_BLOCK    128        // columns per workgroup tile
#define N_BLOCK    128        // rows per workgroup tile (8 waves x 16)
#define M_CHUNKS   (MTOT / M_BLOCK)   // 512
#define N_CHUNKS   (NKP / N_BLOCK)    // 8
#define LDS_STRIDE 264        // halves per LDS B row (padded vs 256)
#define INV_TEMP   100.0f     // 1 / softmax_temp

typedef __attribute__((ext_vector_type(16))) _Float16 v16h;
typedef __attribute__((ext_vector_type(8)))  float    v8f;

union FragU { v16h h; float4 f[2]; };

// ---------------------------------------------------------------------------
// Kernel 1: normalize target keypoint descriptors -> f16 A matrix [b][n][KPAD]
// ---------------------------------------------------------------------------
__global__ void norm_tgt_kernel(const float* __restrict__ kd, _Float16* __restrict__ A) {
    int i = blockIdx.x * blockDim.x + threadIdx.x;
    if (i >= BATCH * NKP) return;
    int b = i >> 10, n = i & (NKP - 1);
    // tgt frame = b*WSIZE + 1
    const float* base = kd + (size_t)(b * WSIZE + 1) * CDESC * NKP + n;
    float ss = 0.f;
    for (int c = 0; c < CDESC; ++c) { float v = base[(size_t)c * NKP]; ss += v * v; }
    float inv = 1.0f / fmaxf(sqrtf(ss), 1e-12f);
    _Float16* arow = A + (size_t)(b * NKP + n) * KPAD;
    for (int c = 0; c < CDESC; ++c) arow[c] = (_Float16)(base[(size_t)c * NKP] * inv);
    for (int c = CDESC; c < KPAD; ++c) arow[c] = (_Float16)0.0f;
}

// ---------------------------------------------------------------------------
// Kernel 2: inverse L2 norms of src dense descriptor columns [b][m]
// ---------------------------------------------------------------------------
__global__ void norm_src_kernel(const float* __restrict__ dd, float* __restrict__ rnorm) {
    int i = blockIdx.x * blockDim.x + threadIdx.x;
    if (i >= BATCH * MTOT) return;
    int b = i >> 16, m = i & (MTOT - 1);
    const float* base = dd + (size_t)(b * WSIZE) * CDESC * MTOT + m;  // src frame = b*WSIZE
    float ss = 0.f;
    for (int c = 0; c < CDESC; ++c) { float v = base[(size_t)c * MTOT]; ss += v * v; }
    rnorm[i] = 1.0f / fmaxf(sqrtf(ss), 1e-12f);
}

// ---------------------------------------------------------------------------
// Kernel 3: WMMA GEMM (tgt x src) fused with streaming-softmax partials.
// grid = (M_CHUNKS, N_CHUNKS, BATCH), block = 256 threads (8 waves of 32).
// One 128x128 output tile per workgroup.  B tile staged (normalized f16,
// transposed [col][k]) in LDS via vectorized b128 global loads.  A fragments
// ping-pong between two registers (load one while WMMAs consume the other)
// so there is no copy and no WMMA->VALU WAR hazard NOPs.
// Partial per (b, n, mChunk): float4{rowmax, sum_e, sum_e*u, sum_e*v}.
// ---------------------------------------------------------------------------
__global__ void gemm_softmax_partial_kernel(const float* __restrict__ dd,
                                            const float* __restrict__ rnorm,
                                            const _Float16* __restrict__ A,
                                            float4* __restrict__ partial) {
    extern __shared__ _Float16 ldsB[];
    const int mChunk = blockIdx.x;
    const int nIter  = blockIdx.y;
    const int b      = blockIdx.z;
    const int tid    = threadIdx.x;
    const int lane   = tid & 31;
    const int wid    = tid >> 5;
    const int m0     = mChunk * M_BLOCK;
    const size_t frameOff = (size_t)(b * WSIZE) * CDESC * MTOT;

    // ---- stage normalized f16 B tile (transposed: [col][k]) ----
    // Each thread owns 4 consecutive m (one b128 load) for c = cGrp (mod 8).
    {
        const int mi4  = (tid & 31) * 4;   // 0..124
        const int cGrp = tid >> 5;         // 0..7
        const float4 rn4 = *(const float4*)(rnorm + b * MTOT + m0 + mi4);
        for (int c = cGrp; c < CDESC; c += 8) {
            float4 v = *(const float4*)(dd + frameOff + (size_t)c * MTOT + m0 + mi4);
            ldsB[(mi4 + 0) * LDS_STRIDE + c] = (_Float16)(v.x * rn4.x);
            ldsB[(mi4 + 1) * LDS_STRIDE + c] = (_Float16)(v.y * rn4.y);
            ldsB[(mi4 + 2) * LDS_STRIDE + c] = (_Float16)(v.z * rn4.z);
            ldsB[(mi4 + 3) * LDS_STRIDE + c] = (_Float16)(v.w * rn4.w);
        }
        if (tid < M_BLOCK) {
            const int mi = tid;
            for (int c = CDESC; c < KPAD; ++c) ldsB[mi * LDS_STRIDE + c] = (_Float16)0.0f;
        }
    }
    __syncthreads();

    const int nBase = nIter * N_BLOCK + wid * 16;
    v8f zero = {};
    v8f acc[8];
#pragma unroll
    for (int t = 0; t < 8; ++t) acc[t] = zero;

    const _Float16* aRow =
        A + (size_t)(b * NKP + nBase + (lane & 15)) * KPAD;
    const int ka = (lane & 16) ? 8 : 0;
    const int kbSel = (lane & 16) ? 8 : 0;
    const int colSel = (lane & 15);

    // prime the A-fragment pipeline (kk = 0) into a0
    FragU a0, a1;
    a0.f[0] = *(const float4*)(aRow + ka);
    a0.f[1] = *(const float4*)(aRow + ka + 16);

#pragma unroll 1
    for (int kk = 0; kk < KPAD / 32; kk += 2) {
        // load A fragment for kk+1 into a1 while WMMAs consume a0
        {
            const _Float16* ap = aRow + ka + (kk + 1) * 32;
            a1.f[0] = *(const float4*)(ap);
            a1.f[1] = *(const float4*)(ap + 16);
        }
        {
            const int kb = kk * 32 + kbSel;
#pragma unroll
            for (int t = 0; t < 8; ++t) {
                FragU bf;
                const _Float16* bp = &ldsB[(t * 16 + colSel) * LDS_STRIDE + kb];
                bf.f[0] = *(const float4*)bp;
                bf.f[1] = *(const float4*)(bp + 16);
                acc[t] = __builtin_amdgcn_wmma_f32_16x16x32_f16(
                    false, a0.h, false, bf.h, (short)0, acc[t], false, false);
            }
        }
        // load A fragment for kk+2 into a0 while WMMAs consume a1
        // (kk==6 reads one fragment past the consumed region of A; this stays
        //  inside the mapped workspace and is never used)
        {
            const _Float16* ap = aRow + ka + (kk + 2) * 32;
            a0.f[0] = *(const float4*)(ap);
            a0.f[1] = *(const float4*)(ap + 16);
        }
        {
            const int kb = (kk + 1) * 32 + kbSel;
#pragma unroll
            for (int t = 0; t < 8; ++t) {
                FragU bf;
                const _Float16* bp = &ldsB[(t * 16 + colSel) * LDS_STRIDE + kb];
                bf.f[0] = *(const float4*)bp;
                bf.f[1] = *(const float4*)(bp + 16);
                acc[t] = __builtin_amdgcn_wmma_f32_16x16x32_f16(
                    false, a1.h, false, bf.h, (short)0, acc[t], false, false);
            }
        }
    }

    // ---- epilogue: per-row streaming softmax partials over this M chunk ----
    // C layout: acc[t] element r -> row (r + (lane>=16 ? 8:0)), col t*16 + (lane&15)
#pragma unroll
    for (int r = 0; r < 8; ++r) {
        float vals[8];
        float mx = -3.402823466e38f;
#pragma unroll
        for (int t = 0; t < 8; ++t) {
            vals[t] = (float)acc[t][r] * INV_TEMP;
            mx = fmaxf(mx, vals[t]);
        }
        for (int off = 1; off < 16; off <<= 1)
            mx = fmaxf(mx, __shfl_xor(mx, off, 16));
        float s = 0.f, su = 0.f, sv = 0.f;
#pragma unroll
        for (int t = 0; t < 8; ++t) {
            float e = __expf(vals[t] - mx);
            int mg = m0 + t * 16 + colSel;
            s  += e;
            su += e * (float)(mg & (WIDTH - 1));   // u = m % 256
            sv += e * (float)(mg >> 8);            // v = m / 256
        }
        for (int off = 1; off < 16; off <<= 1) {
            s  += __shfl_xor(s,  off, 16);
            su += __shfl_xor(su, off, 16);
            sv += __shfl_xor(sv, off, 16);
        }
        if ((lane & 15) == 0) {
            int row = nBase + r + ((lane & 16) ? 8 : 0);
            partial[(size_t)(b * NKP + row) * M_CHUNKS + mChunk] =
                make_float4(mx, s, su, sv);
        }
    }
}

// ---------------------------------------------------------------------------
// Kernel 4: merge 512 partials per (b,n), emit pseudo coords + bilinear time.
// ---------------------------------------------------------------------------
__device__ __forceinline__ float4 smax_merge(float4 a, float4 c) {
    float m  = fmaxf(a.x, c.x);
    float f1 = __expf(a.x - m), f2 = __expf(c.x - m);
    return make_float4(m, a.y * f1 + c.y * f2, a.z * f1 + c.z * f2, a.w * f1 + c.w * f2);
}

__global__ void finalize_kernel(const float4* __restrict__ partial,
                                const float* __restrict__ times,
                                float* __restrict__ out) {
    __shared__ float4 red[128];
    const int b = blockIdx.x >> 10;
    const int n = blockIdx.x & (NKP - 1);
    const int tid = threadIdx.x;   // 0..127
    const float4* p = partial + (size_t)(b * NKP + n) * M_CHUNKS;
    float4 acc = p[tid];
    for (int j = 1; j < M_CHUNKS / 128; ++j) acc = smax_merge(acc, p[tid + j * 128]);
    red[tid] = acc;
    __syncthreads();
    for (int s = 64; s > 0; s >>= 1) {
        if (tid < s) red[tid] = smax_merge(red[tid], red[tid + s]);
        __syncthreads();
    }
    if (tid == 0) {
        float4 t = red[0];
        float u = t.z / t.y, v = t.w / t.y;
        size_t ci = (size_t)(b * NKP + n) * 2;
        out[ci + 0] = u;
        out[ci + 1] = v;
        // align_corners bilinear sample of src times_img (frame b*WSIZE)
        const float* img = times + (size_t)(b * WSIZE) * MTOT;
        float u0 = fminf(fmaxf(floorf(u), 0.f), (float)(WIDTH - 1));
        float v0 = fminf(fmaxf(floorf(v), 0.f), (float)(WIDTH - 1));
        float u1 = fminf(u0 + 1.f, (float)(WIDTH - 1));
        float v1 = fminf(v0 + 1.f, (float)(WIDTH - 1));
        float au = u - u0, av = v - v0;
        int u0i = (int)u0, u1i = (int)u1, v0i = (int)v0, v1i = (int)v1;
        float val = img[v0i * WIDTH + u0i] * (1.f - au) * (1.f - av)
                  + img[v0i * WIDTH + u1i] * au * (1.f - av)
                  + img[v1i * WIDTH + u0i] * (1.f - au) * av
                  + img[v1i * WIDTH + u1i] * au * av;
        out[4096 + 2048 + 2 + b * NKP + n] = val;   // pseudo_times slot
    }
}

// ---------------------------------------------------------------------------
// Kernel 5: copy keypoint_scores[tgt_ids] and tgt_ids into the output blob.
// ---------------------------------------------------------------------------
__global__ void copy_misc_kernel(const float* __restrict__ ks, float* __restrict__ out) {
    int i = blockIdx.x * blockDim.x + threadIdx.x;
    if (i < BATCH * NKP) {
        int b = i >> 10, n = i & (NKP - 1);
        out[4096 + i] = ks[(size_t)(b * WSIZE + 1) * NKP + n];
    }
    if (i == 0) { out[4096 + 2048 + 0] = 1.0f; out[4096 + 2048 + 1] = 3.0f; }
}

// ---------------------------------------------------------------------------
// Launch.  Inputs (setup_inputs order):
//   0: keypoint_scores [4,1,1024] f32
//   1: keypoint_desc   [4,248,1024] f32
//   2: desc_dense      [4,248,256,256] f32
//   3: times_img       [4,1,256,256] f32
// Output blob (8194 f32): coords[4096] | scores[2048] | tgt_ids[2] | times[2048]
// Workspace: A f16 (1 MB) | rnorm f32 (512 KB) | partials float4 (16 MB)
// ---------------------------------------------------------------------------
extern "C" void kernel_launch(void* const* d_in, const int* in_sizes, int n_in,
                              void* d_out, int out_size, void* d_ws, size_t ws_size,
                              hipStream_t stream) {
    const float* ks = (const float*)d_in[0];
    const float* kd = (const float*)d_in[1];
    const float* dd = (const float*)d_in[2];
    const float* ti = (const float*)d_in[3];
    float* out = (float*)d_out;

    char* ws = (char*)d_ws;
    _Float16* A      = (_Float16*)ws;                               // 1 MB
    float*    rnorm  = (float*)(ws + (1u << 20));                   // 512 KB
    float4*   part   = (float4*)(ws + (1u << 20) + (1u << 19));     // 16 MB

    norm_tgt_kernel<<<(BATCH * NKP + 255) / 256, 256, 0, stream>>>(kd, A);
    norm_src_kernel<<<(BATCH * MTOT + 255) / 256, 256, 0, stream>>>(dd, rnorm);

    dim3 grid(M_CHUNKS, N_CHUNKS, BATCH);
    size_t ldsBytes = (size_t)M_BLOCK * LDS_STRIDE * sizeof(_Float16);  // 67584 B
    gemm_softmax_partial_kernel<<<grid, 256, ldsBytes, stream>>>(dd, rnorm, A, part);

    finalize_kernel<<<BATCH * NKP, 128, 0, stream>>>(part, ti, out);
    copy_misc_kernel<<<(BATCH * NKP + 255) / 256, 256, 0, stream>>>(ks, out);
}
